// DecoderWithAttentionNoConcatEncoder_50989851738706
// MI455X (gfx1250) — matrix operand order repaired and checked by reading, not compiled, not run on hardware
//
#include <hip/hip_runtime.h>
#include <hip/hip_bf16.h>
#include <math.h>

#define FEAT  2048
#define HID   512
#define EMB   512
#define NF    512
#define VOCAB 10000
#define BB    64
#define PP    196
#define LL    22
#define TT    21
#define XIN   (EMB + FEAT)   // 2560
#define G4    (4 * HID)      // 2048

typedef __attribute__((ext_vector_type(16))) _Float16 v16h;
typedef __attribute__((ext_vector_type(8)))  _Float16 v8h;
typedef __attribute__((ext_vector_type(8)))  float    v8f;

// ---------------------------------------------------------------------------
// WMMA GEMM with N-register-blocking:
//   C[M,N](f32) = act( A[M,K](f16) * W[N,K]^T(f16) + bias )
// One wave computes a 16 x (16*NB) strip: the A fragment is loaded once per
// K-step and reused across NB column tiles (v_wmma_f32_16x16x32_f16 each).
// Caller guarantees (N/16) % NB == 0, M % 16 == 0, K % 32 == 0.
// act: 0=none 1=tanh 2=sigmoid.  accum: read existing C as the accumulator.
// ---------------------------------------------------------------------------
template <int NB>
__global__ void k_gemm(const _Float16* __restrict__ A, int lda,
                       const _Float16* __restrict__ W, int ldw,
                       const float* __restrict__ bias,
                       float* __restrict__ C, int ldc,
                       int M, int N, int K, int act, int accum)
{
    const int wave    = blockIdx.x * (blockDim.x >> 5) + (threadIdx.x >> 5);
    const int groupsN = (N >> 4) / NB;
    const int nGroups = (M >> 4) * groupsN;
    if (wave >= nGroups) return;              // wave-uniform: EXEC all-ones inside
    const int tm   = (wave / groupsN) << 4;
    const int tn0  = (wave % groupsN) * (16 * NB);
    const int lane = threadIdx.x & 31;
    const int half = lane >> 4;
    const int l16  = lane & 15;

    v8f c[NB];
    if (accum) {
#pragma unroll
        for (int j = 0; j < NB; ++j)
#pragma unroll
            for (int v = 0; v < 8; ++v)
                c[j][v] = C[(size_t)(tm + (half << 3) + v) * ldc + tn0 + j * 16 + l16];
    } else {
#pragma unroll
        for (int j = 0; j < NB; ++j) {
            const float b0 = bias ? bias[tn0 + j * 16 + l16] : 0.0f;
#pragma unroll
            for (int v = 0; v < 8; ++v) c[j][v] = b0;
        }
    }

    // A fragment: row = l16; K element e -> k0 + 16*half + e
    const _Float16* __restrict__ arow = A + (size_t)(tm + l16) * lda;
    // B fragment j: column (=row of row-major W) = tn0+16j+l16; K = k0+16*half+e
    const _Float16* __restrict__ wrow = W + (size_t)(tn0 + l16) * ldw;

    for (int k0 = 0; k0 < K; k0 += 32) {
        union { v16h v; v8h h[2]; } a;
        a.v = *(const v16h*)(arow + k0 + (half << 4));
#pragma unroll
        for (int j = 0; j < NB; ++j) {
            v16h b = *(const v16h*)(wrow + (size_t)j * 16 * ldw + k0 + (half << 4));
            c[j] = __builtin_amdgcn_wmma_f32_16x16x32_f16(
                       /*neg_a=*/false, a.v, /*neg_b=*/false, b,
                       /*c_mod=*/(short)0, c[j], /*reuse_a=*/false, /*reuse_b=*/false);
        }
    }

#pragma unroll
    for (int j = 0; j < NB; ++j)
#pragma unroll
        for (int v = 0; v < 8; ++v) {
            float x = c[j][v];
            if (act == 1)      x = tanhf(x);
            else if (act == 2) x = 1.0f / (1.0f + __expf(-x));
            C[(size_t)(tm + (half << 3) + v) * ldc + tn0 + j * 16 + l16] = x;
        }
}

// ---------------------------------------------------------------------------
// Per-batch reductions: aesthetic mean+max -> first_in f16 (B,2*FEAT),
// factual mean -> fm16 (B,FEAT).
// ---------------------------------------------------------------------------
__global__ void k_prep(const float* __restrict__ fact, const float* __restrict__ aest,
                       _Float16* __restrict__ first_in, _Float16* __restrict__ fm16)
{
    const int b = blockIdx.x;
    for (int f = threadIdx.x; f < FEAT; f += blockDim.x) {
        const float* ap = aest + (size_t)b * PP * FEAT + f;
        const float* fp = fact + (size_t)b * PP * FEAT + f;
        float s = 0.f, mx = -INFINITY, fs = 0.f;
        for (int p = 0; p < PP; ++p) {
            const float av = ap[(size_t)p * FEAT];
            s += av; mx = fmaxf(mx, av);
            fs += fp[(size_t)p * FEAT];
        }
        first_in[(size_t)b * (2 * FEAT) + f]        = (_Float16)(s / PP);
        first_in[(size_t)b * (2 * FEAT) + FEAT + f] = (_Float16)mx;
        fm16[(size_t)b * FEAT + f]                  = (_Float16)(fs / PP);
    }
}

__global__ void k_cvt(const float* __restrict__ in, _Float16* __restrict__ out, long n)
{
    long i = (long)blockIdx.x * blockDim.x + threadIdx.x;
    const long stride = (long)gridDim.x * blockDim.x;
    for (; i < n; i += stride) out[i] = (_Float16)in[i];
}

__global__ void k_bias_sum(const float* __restrict__ a, const float* __restrict__ b,
                           float* __restrict__ o, int n)
{
    const int i = blockIdx.x * blockDim.x + threadIdx.x;
    if (i < n) o[i] = a[i] + b[i];
}

// embv[t,b,:] : t==0 -> tanh'ed "first" vector; t>=1 -> emb_W[gt[b,t]]
__global__ void k_embed(const float* __restrict__ first, const float* __restrict__ embW,
                        const int* __restrict__ gt, _Float16* __restrict__ embv)
{
    const int t = blockIdx.x / BB;
    const int b = blockIdx.x % BB;
    _Float16* dst = embv + ((size_t)t * BB + b) * EMB;
    const float* src;
    if (t == 0) src = first + (size_t)b * EMB;
    else        src = embW + (size_t)gt[b * LL + t] * EMB;
    for (int j = threadIdx.x; j < EMB; j += blockDim.x) dst[j] = (_Float16)src[j];
}

// ---------------------------------------------------------------------------
// Attention step: e = v . relu(fa + ha), softmax over P, ctx = sum_p a_p*feat,
// xin = [emb_t, gate*ctx] (f16), write masked alpha to output.
// One block per batch element.
// ---------------------------------------------------------------------------
__global__ void k_attn(const float* __restrict__ fa,  const float* __restrict__ ha,
                       const float* __restrict__ fact, const float* __restrict__ vW,
                       const float* __restrict__ vb,  const float* __restrict__ gate,
                       const _Float16* __restrict__ embv, const int* __restrict__ clen,
                       _Float16* __restrict__ xin, float* __restrict__ alpha_out, int t)
{
    __shared__ float e[PP];
    __shared__ float red[2];
    const int b    = blockIdx.x;
    const int tid  = threadIdx.x;
    const int wid  = tid >> 5, lane = tid & 31, nw = blockDim.x >> 5;
    const float* fab = fa + (size_t)b * PP * NF;
    const float* hab = ha + (size_t)b * NF;

    for (int p = wid; p < PP; p += nw) {
        float s = 0.f;
        const float* fr = fab + (size_t)p * NF;
        for (int j = lane; j < NF; j += 32)
            s += fmaxf(fr[j] + hab[j], 0.f) * vW[j];
        for (int off = 16; off; off >>= 1) s += __shfl_down(s, off, 32);
        if (lane == 0) e[p] = s + vb[0];
    }
    __syncthreads();
    if (tid == 0) {
        float mx = -INFINITY;
        for (int p = 0; p < PP; ++p) mx = fmaxf(mx, e[p]);
        float sum = 0.f;
        for (int p = 0; p < PP; ++p) sum += __expf(e[p] - mx);
        red[0] = mx; red[1] = sum;
    }
    __syncthreads();
    const bool act = t < (clen[b] - 1);
    float a = 0.f;
    if (tid < PP) a = __expf(e[tid] - red[0]) / red[1];
    __syncthreads();
    if (tid < PP) {
        e[tid] = a;
        alpha_out[((size_t)b * TT + t) * PP + tid] = act ? a : 0.f;
    }
    __syncthreads();

    const float* fb2 = fact + (size_t)b * PP * FEAT;
    for (int f = tid; f < FEAT; f += blockDim.x) {
        float acc = 0.f;
        for (int p = 0; p < PP; ++p) acc += e[p] * fb2[(size_t)p * FEAT + f];
        xin[(size_t)b * XIN + EMB + f] = (_Float16)(gate[(size_t)b * FEAT + f] * acc);
    }
    for (int j = tid; j < EMB; j += blockDim.x)
        xin[(size_t)b * XIN + j] = embv[((size_t)t * BB + b) * EMB + j];
}

// LSTM pointwise + masked state update
__global__ void k_lstm(const float* __restrict__ g, const int* __restrict__ clen,
                       float* __restrict__ h, float* __restrict__ c,
                       _Float16* __restrict__ h16, _Float16* __restrict__ hnew16, int t)
{
    const int i = blockIdx.x * blockDim.x + threadIdx.x;
    if (i >= BB * HID) return;
    const int b = i / HID, j = i % HID;
    const float* gb = g + (size_t)b * G4;
    const float ig = 1.f / (1.f + __expf(-gb[j]));
    const float fg = 1.f / (1.f + __expf(-gb[HID + j]));
    const float gg = tanhf(gb[2 * HID + j]);
    const float og = 1.f / (1.f + __expf(-gb[3 * HID + j]));
    const float cold = c[i], hold = h[i];
    const float cn = fg * cold + ig * gg;
    const float hn = og * tanhf(cn);
    const bool  act = t < (clen[b] - 1);
    const float h2 = act ? hn : hold;
    const float c2 = act ? cn : cold;
    h[i] = h2; c[i] = c2;
    h16[i]    = (_Float16)h2;
    hnew16[i] = (_Float16)hn;
}

__global__ void k_maskpred(float* __restrict__ preds, const int* __restrict__ clen, int t)
{
    const int i = blockIdx.x * blockDim.x + threadIdx.x;
    if (i >= BB * VOCAB) return;
    const int b = i / VOCAB, v = i % VOCAB;
    if (t >= clen[b] - 1) preds[((size_t)b * TT + t) * VOCAB + v] = 0.f;
}

// ---------------------------------------------------------------------------
// Host side
// ---------------------------------------------------------------------------
static inline void gemm(hipStream_t s, const _Float16* A, int lda,
                        const _Float16* W, int ldw, const float* bias,
                        float* C, int ldc, int M, int N, int K, int act, int accum)
{
    const int tilesN = N / 16;
    if (tilesN % 4 == 0) {
        const int groups = (M / 16) * (tilesN / 4);
        k_gemm<4><<<(groups + 7) / 8, 256, 0, s>>>(A, lda, W, ldw, bias, C, ldc,
                                                   M, N, K, act, accum);
    } else if (tilesN % 5 == 0) {
        const int groups = (M / 16) * (tilesN / 5);
        k_gemm<5><<<(groups + 7) / 8, 256, 0, s>>>(A, lda, W, ldw, bias, C, ldc,
                                                   M, N, K, act, accum);
    } else {
        const int groups = (M / 16) * tilesN;
        k_gemm<1><<<(groups + 7) / 8, 256, 0, s>>>(A, lda, W, ldw, bias, C, ldc,
                                                   M, N, K, act, accum);
    }
}

static inline void cvt(hipStream_t s, const float* in, _Float16* out, long n)
{
    int blocks = (int)((n + 255) / 256);
    if (blocks > 2048) blocks = 2048;
    k_cvt<<<blocks, 256, 0, s>>>(in, out, n);
}

extern "C" void kernel_launch(void* const* d_in, const int* in_sizes, int n_in,
                              void* d_out, int out_size, void* d_ws, size_t ws_size,
                              hipStream_t stream)
{
    (void)in_sizes; (void)n_in; (void)out_size; (void)ws_size;
    const float* factual = (const float*)d_in[0];
    const float* aest    = (const float*)d_in[1];
    const int*   gt      = (const int*)  d_in[2];
    const int*   clen    = (const int*)  d_in[3];
    const float* embW    = (const float*)d_in[4];
    const float* fa_W    = (const float*)d_in[5];
    const float* fa_b    = (const float*)d_in[6];
    const float* ha_W    = (const float*)d_in[7];
    const float* ha_b    = (const float*)d_in[8];
    const float* v_W     = (const float*)d_in[9];
    const float* v_b     = (const float*)d_in[10];
    const float* ih_W    = (const float*)d_in[11];
    const float* ih_b    = (const float*)d_in[12];
    const float* ic_W    = (const float*)d_in[13];
    const float* ic_b    = (const float*)d_in[14];
    const float* if_W    = (const float*)d_in[15];
    const float* if_b    = (const float*)d_in[16];
    const float* fb_W    = (const float*)d_in[17];
    const float* fb_b    = (const float*)d_in[18];
    const float* Wih     = (const float*)d_in[19];
    const float* Whh     = (const float*)d_in[20];
    const float* bih     = (const float*)d_in[21];
    const float* bhh     = (const float*)d_in[22];
    const float* fc_W    = (const float*)d_in[23];
    const float* fc_b    = (const float*)d_in[24];

    float* preds_out = (float*)d_out;                                  // (B,T,VOCAB)
    float* alpha_out = (float*)d_out + (size_t)BB * TT * VOCAB;        // (B,T,P)

    // --- workspace carve-up (256B aligned) ---
    char* p = (char*)d_ws;
    auto take = [&](size_t bytes) {
        char* r = p;
        p += (bytes + 255) & ~(size_t)255;
        return (void*)r;
    };
    _Float16* fact16  = (_Float16*)take((size_t)BB * PP * FEAT * 2);
    _Float16* w_fa    = (_Float16*)take((size_t)NF * FEAT * 2);
    _Float16* w_ha    = (_Float16*)take((size_t)NF * HID * 2);
    _Float16* w_ih    = (_Float16*)take((size_t)HID * FEAT * 2);
    _Float16* w_ic    = (_Float16*)take((size_t)HID * FEAT * 2);
    _Float16* w_if    = (_Float16*)take((size_t)EMB * 2 * FEAT * 2);
    _Float16* w_fb    = (_Float16*)take((size_t)FEAT * HID * 2);
    _Float16* w_wih   = (_Float16*)take((size_t)G4 * XIN * 2);
    _Float16* w_whh   = (_Float16*)take((size_t)G4 * HID * 2);
    _Float16* w_fc    = (_Float16*)take((size_t)VOCAB * HID * 2);
    _Float16* first16 = (_Float16*)take((size_t)BB * 2 * FEAT * 2);
    _Float16* fm16    = (_Float16*)take((size_t)BB * FEAT * 2);
    _Float16* embv    = (_Float16*)take((size_t)TT * BB * EMB * 2);
    _Float16* xin16   = (_Float16*)take((size_t)BB * XIN * 2);
    _Float16* h16     = (_Float16*)take((size_t)BB * HID * 2);
    _Float16* hnew16  = (_Float16*)take((size_t)BB * HID * 2);
    float* fa_out  = (float*)take((size_t)BB * PP * NF * 4);
    float* first_f = (float*)take((size_t)BB * EMB * 4);
    float* h_f     = (float*)take((size_t)BB * HID * 4);
    float* c_f     = (float*)take((size_t)BB * HID * 4);
    float* ha_out  = (float*)take((size_t)BB * NF * 4);
    float* gate_f  = (float*)take((size_t)BB * FEAT * 4);
    float* g_f     = (float*)take((size_t)BB * G4 * 4);
    float* bsum    = (float*)take((size_t)G4 * 4);

    // --- one-time conversions / reductions ---
    cvt(stream, factual, fact16, (long)BB * PP * FEAT);
    cvt(stream, fa_W, w_fa, (long)NF * FEAT);
    cvt(stream, ha_W, w_ha, (long)NF * HID);
    cvt(stream, ih_W, w_ih, (long)HID * FEAT);
    cvt(stream, ic_W, w_ic, (long)HID * FEAT);
    cvt(stream, if_W, w_if, (long)EMB * 2 * FEAT);
    cvt(stream, fb_W, w_fb, (long)FEAT * HID);
    cvt(stream, Wih,  w_wih, (long)G4 * XIN);
    cvt(stream, Whh,  w_whh, (long)G4 * HID);
    cvt(stream, fc_W, w_fc, (long)VOCAB * HID);
    k_prep<<<BB, 256, 0, stream>>>(factual, aest, first16, fm16);
    k_bias_sum<<<(G4 + 255) / 256, 256, 0, stream>>>(bih, bhh, bsum, G4);

    // first = tanh([avg,max] @ if_W.T + if_b)
    gemm(stream, first16, 2 * FEAT, w_if, 2 * FEAT, if_b, first_f, EMB,
         BB, EMB, 2 * FEAT, /*act=*/1, 0);
    // h0, c0
    gemm(stream, fm16, FEAT, w_ih, FEAT, ih_b, h_f, HID, BB, HID, FEAT, 1, 0);
    gemm(stream, fm16, FEAT, w_ic, FEAT, ic_b, c_f, HID, BB, HID, FEAT, 1, 0);
    cvt(stream, h_f, h16, (long)BB * HID);
    // fa = factual @ fa_W.T + fa_b   (12544 x 2048 x 512)
    gemm(stream, fact16, FEAT, w_fa, FEAT, fa_b, fa_out, NF,
         BB * PP, NF, FEAT, 0, 0);
    // embedding sequence (t=0 uses `first`)
    k_embed<<<TT * BB, 256, 0, stream>>>(first_f, embW, gt, embv);

    // --- sequential scan, T = 21 steps ---
    for (int t = 0; t < TT; ++t) {
        // ha = h @ ha_W.T + ha_b
        gemm(stream, h16, HID, w_ha, HID, ha_b, ha_out, NF, BB, NF, HID, 0, 0);
        // gate = sigmoid(h @ fb_W.T + fb_b)
        gemm(stream, h16, HID, w_fb, HID, fb_b, gate_f, FEAT, BB, FEAT, HID, 2, 0);
        // attention softmax + context + xin assembly
        k_attn<<<BB, 256, 0, stream>>>(fa_out, ha_out, factual, v_W, v_b,
                                       gate_f, embv, clen, xin16, alpha_out, t);
        // g = xin @ Wih.T + (bih+bhh)  then  += h @ Whh.T
        gemm(stream, xin16, XIN, w_wih, XIN, bsum, g_f, G4, BB, G4, XIN, 0, 0);
        gemm(stream, h16, HID, w_whh, HID, nullptr, g_f, G4, BB, G4, HID, 0, 1);
        // LSTM pointwise + masked state update
        k_lstm<<<(BB * HID + 255) / 256, 256, 0, stream>>>(g_f, clen, h_f, c_f,
                                                           h16, hnew16, t);
        // preds = h_new @ fc_W.T + fc_b  (written straight into d_out)
        gemm(stream, hnew16, HID, w_fc, HID, fc_b,
             preds_out + (size_t)t * VOCAB, TT * VOCAB, BB, VOCAB, HID, 0, 0);
        k_maskpred<<<(BB * VOCAB + 255) / 256, 256, 0, stream>>>(preds_out, clen, t);
    }
}
